// GraphPredictionPropagator_18949395710195
// MI455X (gfx1250) — matrix operand order repaired and checked by reading, not compiled, not run on hardware
//
#include <hip/hip_runtime.h>
#include <hip/hip_bf16.h>

#define T_DIM 4
#define H_DIM 64
#define NEG_SLOPE 0.2f
#define LN_EPS 1e-5f

typedef __attribute__((ext_vector_type(16))) _Float16 v16h;
typedef __attribute__((ext_vector_type(8)))  float    v8f;

// ---- ordered-float encoding for atomicMax on floats ----
__device__ __forceinline__ unsigned f2ord(float f) {
  unsigned u = __float_as_uint(f);
  return (u & 0x80000000u) ? ~u : (u | 0x80000000u);
}
__device__ __forceinline__ float ord2f(unsigned u) {
  return (u & 0x80000000u) ? __uint_as_float(u & 0x7fffffffu)
                           : __uint_as_float(~u);
}
#define KEY_NEG_INF 0x007FFFFFu  // f2ord(-inf)

// ---------------- init ----------------
__global__ void fill_u32_kernel(unsigned* __restrict__ p, unsigned v, int n) {
  int i = blockIdx.x * blockDim.x + threadIdx.x;
  int stride = gridDim.x * blockDim.x;
  for (; i < n; i += stride) p[i] = v;
}

// ---------------- fused GEMM + row-dot ----------------
// Hout[r,d] = sum_k X[r,k] * W[k,d]   (K = D = 64)
// srow[r]   = sum_d Hout[r,d] * a[d]
//
// Block = 256 threads (8 waves). W (16 KB, f32) is converted to f16 and staged
// in LDS pre-swizzled into the 8 WMMA B-fragments (4 D-tiles x 2 K-chunks), so
// each wave reads B with contiguous ds_load_b128s. Each wave owns one 16-row
// M-tile, loads its A fragments once, and runs 8 v_wmma_f32_16x16x32_f16 to
// produce all 64 output columns, accumulating the a-dot in registers.
__global__ void gemm_xw_wmma_kernel(const float* __restrict__ X,
                                    const float* __restrict__ W,
                                    const float* __restrict__ avec,
                                    float* __restrict__ Hout,
                                    float* __restrict__ srow) {
  __shared__ _Float16 wlds[8 * 512];   // 8 fragments * 512 halfs = 8 KB

  const int tid  = threadIdx.x;
  const int lane = tid & 31;
  const int wv   = tid >> 5;           // wave 0..7
  const int col  = lane & 15;
  const int hi   = lane >> 4;          // lane half

  // ---- stage W fragments: thread tid builds fragment f = tid>>5 for lane tid&31
  {
    const int f  = tid >> 5;           // fragment 0..7 = dt*2 + kc
    const int kc = f & 1;
    const int dt = f >> 1;
    v16h bfrag;
#pragma unroll
    for (int j = 0; j < 16; ++j) {
      // B k order used here (contiguous, same as A): lanes 0-15 hold K 0-15, lanes 16-31 hold K 16-31
      int kB = j + (hi << 4);
      bfrag[j] = (_Float16)W[(kc * 32 + kB) * H_DIM + dt * 16 + col];
    }
    *(v16h*)&wlds[f * 512 + lane * 16] = bfrag;
  }
  __syncthreads();

  // ---- A fragments for this wave's 16-row M-tile
  const int m0 = (blockIdx.x * 8 + wv) << 4;
  const float* xr = X + (size_t)(m0 + col) * H_DIM;
  v16h a0, a1;
#pragma unroll
  for (int j = 0; j < 16; ++j) {
    // A k order used here (contiguous, same as B): lanes 0-15 hold K 0-15, lanes 16-31 hold K 16-31
    int kA = j + (hi << 4);
    a0[j] = (_Float16)xr[kA];
    a1[j] = (_Float16)xr[32 + kA];
  }

  float dotacc[8];
#pragma unroll
  for (int v = 0; v < 8; ++v) dotacc[v] = 0.f;

#pragma unroll
  for (int dt = 0; dt < 4; ++dt) {
    v16h b0 = *(const v16h*)&wlds[(dt * 2 + 0) * 512 + lane * 16];
    v16h b1 = *(const v16h*)&wlds[(dt * 2 + 1) * 512 + lane * 16];
    v8f c = {};
    c = __builtin_amdgcn_wmma_f32_16x16x32_f16(false, a0, false, b0,
                                               (short)0, c, false, false);
    c = __builtin_amdgcn_wmma_f32_16x16x32_f16(false, a1, false, b1,
                                               (short)0, c, false, false);
    const int d0 = dt << 4;
    const float av = avec[d0 + col];
#pragma unroll
    for (int v = 0; v < 8; ++v) {
      int row = m0 + v + (hi << 3);    // C/D layout: VGPR v -> M = v (+8 upper half)
      Hout[(size_t)row * H_DIM + d0 + col] = c[v];
      dotacc[v] += c[v] * av;
    }
  }

  // reduce dot over the 16 lanes of each half-wave (columns 0..63 gathered
  // across the 4 D-tiles already accumulated in dotacc)
#pragma unroll
  for (int off = 8; off > 0; off >>= 1) {
#pragma unroll
    for (int v = 0; v < 8; ++v) dotacc[v] += __shfl_xor(dotacc[v], off, 32);
  }
  if (col == 0) {
#pragma unroll
    for (int v = 0; v < 8; ++v) srow[m0 + v + (hi << 3)] = dotacc[v];
  }
}

// ---------------- edge pass 1: segment max of leaky_relu(s[src]) over dst ----------------
__global__ void edge_max_kernel(const int* __restrict__ src, const int* __restrict__ dst,
                                const float* __restrict__ s,
                                unsigned* __restrict__ mkey, int E) {
  int e = blockIdx.x * blockDim.x + threadIdx.x;
  if (e >= E) return;
  int sn = src[e], dn = dst[e];
  float4 sv = ((const float4*)s)[sn];
  float l0 = sv.x > 0.f ? sv.x : NEG_SLOPE * sv.x;
  float l1 = sv.y > 0.f ? sv.y : NEG_SLOPE * sv.y;
  float l2 = sv.z > 0.f ? sv.z : NEG_SLOPE * sv.z;
  float l3 = sv.w > 0.f ? sv.w : NEG_SLOPE * sv.w;
  unsigned* mk = mkey + dn * T_DIM;
  atomicMax(mk + 0, f2ord(l0));
  atomicMax(mk + 1, f2ord(l1));
  atomicMax(mk + 2, f2ord(l2));
  atomicMax(mk + 3, f2ord(l3));
}

// ---------------- edge pass 2: ex = exp(logit - max); den[dst] += ex ----------------
__global__ void edge_exp_kernel(const int* __restrict__ src, const int* __restrict__ dst,
                                const float* __restrict__ s,
                                const unsigned* __restrict__ mkey,
                                float* __restrict__ den, float* __restrict__ exb, int E) {
  int e = blockIdx.x * blockDim.x + threadIdx.x;
  if (e >= E) return;
  int sn = src[e], dn = dst[e];
  float4 sv = ((const float4*)s)[sn];
  uint4  mk = ((const uint4*)mkey)[dn];
  float l0 = sv.x > 0.f ? sv.x : NEG_SLOPE * sv.x;
  float l1 = sv.y > 0.f ? sv.y : NEG_SLOPE * sv.y;
  float l2 = sv.z > 0.f ? sv.z : NEG_SLOPE * sv.z;
  float l3 = sv.w > 0.f ? sv.w : NEG_SLOPE * sv.w;
  float4 ev;
  ev.x = __expf(l0 - ord2f(mk.x));
  ev.y = __expf(l1 - ord2f(mk.y));
  ev.z = __expf(l2 - ord2f(mk.z));
  ev.w = __expf(l3 - ord2f(mk.w));
  ((float4*)exb)[e] = ev;
  float* dp = den + dn * T_DIM;
  atomicAdd(dp + 0, ev.x);
  atomicAdd(dp + 1, ev.y);
  atomicAdd(dp + 2, ev.z);
  atomicAdd(dp + 3, ev.w);
}

// ---------------- edge pass 3: acc[dst] += 0.5 * alpha * h[src]  (wave per edge) ----------------
__global__ void edge_scatter_kernel(const int* __restrict__ src, const int* __restrict__ dst,
                                    const float* __restrict__ exb, const float* __restrict__ den,
                                    const float* __restrict__ Hrow, float* __restrict__ acc, int E) {
  int lane = threadIdx.x & 31;
  int e = blockIdx.x * (blockDim.x >> 5) + (threadIdx.x >> 5);
  if (e >= E) return;
  int sn = src[e], dn = dst[e];
  float4 exv = ((const float4*)exb)[e];
  float4 dnv = ((const float4*)den)[dn];
  float alpha[T_DIM] = { 0.5f * exv.x / dnv.x, 0.5f * exv.y / dnv.y,
                         0.5f * exv.z / dnv.z, 0.5f * exv.w / dnv.w };
#pragma unroll
  for (int t = 0; t < T_DIM; ++t) {
    const float2 hv = ((const float2*)(Hrow + ((size_t)sn * T_DIM + t) * H_DIM))[lane];
    float* ap = acc + ((size_t)dn * T_DIM + t) * H_DIM + 2 * lane;
    atomicAdd(ap,     alpha[t] * hv.x);
    atomicAdd(ap + 1, alpha[t] * hv.y);
  }
}

// ---------------- residual + LayerNorm (wave per row) ----------------
__global__ void combine_ln_kernel(const float* __restrict__ xin, const float* __restrict__ acc,
                                  const float* __restrict__ g, const float* __restrict__ b,
                                  float* __restrict__ xout, int rows) {
  int lane = threadIdx.x & 31;
  int r = blockIdx.x * (blockDim.x >> 5) + (threadIdx.x >> 5);
  if (r >= rows) return;
  const float2 xv = ((const float2*)(xin + (size_t)r * H_DIM))[lane];
  const float2 av = ((const float2*)(acc + (size_t)r * H_DIM))[lane];
  float2 v; v.x = xv.x + av.x; v.y = xv.y + av.y;
  float sum = v.x + v.y;
  float sq  = v.x * v.x + v.y * v.y;
#pragma unroll
  for (int off = 16; off > 0; off >>= 1) {
    sum += __shfl_xor(sum, off, 32);
    sq  += __shfl_xor(sq,  off, 32);
  }
  float mu  = sum * (1.0f / H_DIM);
  float var = sq * (1.0f / H_DIM) - mu * mu;
  float inv = rsqrtf(var + LN_EPS);
  const float2 gv = ((const float2*)g)[lane];
  const float2 bv = ((const float2*)b)[lane];
  float2 y;
  y.x = (v.x - mu) * inv * gv.x + bv.x;
  y.y = (v.y - mu) * inv * gv.y + bv.y;
  ((float2*)(xout + (size_t)r * H_DIM))[lane] = y;
}

extern "C" void kernel_launch(void* const* d_in, const int* in_sizes, int n_in,
                              void* d_out, int out_size, void* d_ws, size_t ws_size,
                              hipStream_t stream) {
  (void)n_in; (void)out_size; (void)ws_size;
  const float* pred = (const float*)d_in[0];
  const int E    = in_sizes[1] / 2;        // 262144
  const int rows = in_sizes[0] / H_DIM;    // N*T = 65536

  const int* ei[2] = { (const int*)d_in[1], (const int*)d_in[2] };
  const float* Wp[2][2] = { { (const float*)d_in[3],  (const float*)d_in[5]  },
                            { (const float*)d_in[9],  (const float*)d_in[11] } };
  const float* ap[2][2] = { { (const float*)d_in[4],  (const float*)d_in[6]  },
                            { (const float*)d_in[10], (const float*)d_in[12] } };
  const float* gp[2] = { (const float*)d_in[7],  (const float*)d_in[13] };
  const float* bp[2] = { (const float*)d_in[8],  (const float*)d_in[14] };

  const size_t NTH = (size_t)rows * H_DIM;
  float*    xbuf = (float*)d_ws;
  float*    hbuf = xbuf + NTH;
  float*    acc  = hbuf + NTH;
  float*    srow = acc + NTH;
  unsigned* mkey = (unsigned*)(srow + rows);
  float*    den  = (float*)(mkey + rows);
  float*    exb  = den + rows;

  const float* xin = pred;
  for (int l = 0; l < 2; ++l) {
    float* xout = (l == 0) ? xbuf : (float*)d_out;
    fill_u32_kernel<<<2048, 256, 0, stream>>>((unsigned*)acc, 0u, (int)NTH);
    for (int et = 0; et < 2; ++et) {
      // 8 waves/block, one 16-row M-tile per wave covering all 64 columns
      gemm_xw_wmma_kernel<<<rows / 128, 256, 0, stream>>>(xin, Wp[l][et], ap[l][et],
                                                          hbuf, srow);
      fill_u32_kernel<<<256, 256, 0, stream>>>(mkey, KEY_NEG_INF, rows);
      fill_u32_kernel<<<256, 256, 0, stream>>>((unsigned*)den, 0u, rows);
      edge_max_kernel<<<(E + 255) / 256, 256, 0, stream>>>(ei[et], ei[et] + E, srow, mkey, E);
      edge_exp_kernel<<<(E + 255) / 256, 256, 0, stream>>>(ei[et], ei[et] + E, srow, mkey, den, exb, E);
      edge_scatter_kernel<<<(E + 7) / 8, 256, 0, stream>>>(ei[et], ei[et] + E, exb, den, hbuf, acc, E);
    }
    combine_ln_kernel<<<(rows + 7) / 8, 256, 0, stream>>>(xin, acc, gp[l], bp[l], xout, rows);
    xin = xbuf;
  }
}
